// MultiHeadAttention_66709432041511
// MI455X (gfx1250) — matrix operand order repaired and checked by reading, not compiled, not run on hardware
//
#include <hip/hip_runtime.h>
#include <cstdint>
#include <cstddef>

#ifndef __has_builtin
#define __has_builtin(x) 0
#endif

typedef __attribute__((ext_vector_type(16))) _Float16 v16h;
typedef __attribute__((ext_vector_type(8)))  _Float16 v8h;
typedef __attribute__((ext_vector_type(8)))  float    v8f;
typedef __attribute__((ext_vector_type(4)))  unsigned int v4u;
typedef __attribute__((ext_vector_type(8)))  int v8i_t;
typedef __attribute__((ext_vector_type(4)))  int v4i_t;
typedef int v4i_vs __attribute__((vector_size(16)));  // matches async builtin param

#define BATCH  4
#define SEQ    2048
#define DMODEL 1024
#define NHEAD  16
#define HDIM   64
#define MROWS  (BATCH * SEQ)   // 8192

// ---------------------------------------------------------------------------
// CDNA5 async global->LDS copy (ASYNCcnt path), with plain-copy fallback.
// Builtin signature (verified by hipcc diagnostic): (v4i AS1*, v4i AS3*, Ii, Ii)
// ---------------------------------------------------------------------------
#if __has_builtin(__builtin_amdgcn_global_load_async_to_lds_b128)
#define HAVE_ASYNC 1
#else
#define HAVE_ASYNC 0
#endif

#if __has_builtin(__builtin_amdgcn_tensor_load_to_lds)
#define HAVE_TDM 1
#else
#define HAVE_TDM 0
#endif

__device__ __forceinline__ void async_copy16(void* lds, const void* g) {
#if HAVE_ASYNC
  __builtin_amdgcn_global_load_async_to_lds_b128(
      (__attribute__((address_space(1))) v4i_vs*)(uintptr_t)g,
      (__attribute__((address_space(3))) v4i_vs*)(uintptr_t)lds, 0, 0);
#else
  *(v4u*)lds = *(const v4u*)g;
#endif
}

__device__ __forceinline__ void wait_async_copies() {
#if HAVE_ASYNC
#if __has_builtin(__builtin_amdgcn_s_wait_asynccnt)
  __builtin_amdgcn_s_wait_asynccnt(0);
#else
  asm volatile("s_wait_asynccnt 0" ::: "memory");
#endif
#endif
}

// ---------------------------------------------------------------------------
// Tensor Data Mover: 2D tile (elements are 2 bytes) global -> LDS.
// D# built per CDNA5 ISA ch.8: group0 = {count/lds_addr/global_addr/type},
// group1 = {flags, tensor dims, tile dims, dim0 stride}.
// pad=true inserts 4 DWORDs of LDS padding every 16 DWORDs (64B row -> 80B).
// Toolchain uses the 6-arg builtin: (u32x4, i32x8, i32x4, i32x4, i32x8, cpol).
// ---------------------------------------------------------------------------
#if HAVE_TDM
__device__ __forceinline__ void tdm_load_2d(const void* gsrc, void* ldsDst,
                                            unsigned tileW, unsigned tileH,
                                            unsigned rowStrideElems, bool pad) {
  unsigned long long ga = (unsigned long long)(uintptr_t)gsrc;
  v4u g0;
  g0.x = 1u;                                  // count=1 (valid user descriptor)
  g0.y = (unsigned)(uintptr_t)ldsDst;         // lds_addr (bytes)
  g0.z = (unsigned)ga;                        // global_addr[31:0]
  g0.w = (unsigned)((ga >> 32) & 0x01FFFFFFu) // global_addr[56:32]
         | (2u << 30);                        // type = 2 ("image")
  unsigned flags = (1u << 16);                // data_size = 1 -> 2 bytes
  if (pad) flags |= (1u << 20)                // pad_enable
                  | (3u << 22)                // pad_interval: 16 DWORDs
                  | (3u << 25);               // pad_amount: 4 DWORDs
  const unsigned td0 = 0x40000000u;           // huge tensor dims: no OOB clip
  const unsigned td1 = 0x00100000u;
  v8i_t g1;
  g1[0] = (int)flags;                         // wg_mask=0, flags
  g1[1] = (int)((td0 & 0xFFFFu) << 16);       // tensor_dim0[15:0] @ 63:48
  g1[2] = (int)((td0 >> 16) | ((td1 & 0xFFFFu) << 16));  // dim0 hi | dim1 lo
  g1[3] = (int)((td1 >> 16) | (tileW << 16)); // dim1 hi | tile_dim0
  g1[4] = (int)(tileH & 0xFFFFu);             // tile_dim1 (tile_dim2 = 0)
  g1[5] = (int)rowStrideElems;                // tensor_dim0_stride[31:0]
  g1[6] = 0;                                  // stride hi | dim1_stride lo
  g1[7] = 0;
  v4i_t z4 = (v4i_t){0, 0, 0, 0};
  v8i_t z8 = (v8i_t){0, 0, 0, 0, 0, 0, 0, 0};
  __builtin_amdgcn_tensor_load_to_lds(g0, g1, z4, z4, z8, 0);
}
#endif

__device__ __forceinline__ void wait_tile_stage() {
#if HAVE_TDM
#if __has_builtin(__builtin_amdgcn_s_wait_tensorcnt)
  __builtin_amdgcn_s_wait_tensorcnt(0);
#else
  asm volatile("s_wait_tensorcnt 0" ::: "memory");
#endif
#else
  wait_async_copies();
#endif
}

__device__ __forceinline__ v8f wmma16(v16h a, v16h b, v8f c) {
  // D = A(16x32 f16) * B(32x16 f16) + C(16x16 f32)
  return __builtin_amdgcn_wmma_f32_16x16x32_f16(
      /*neg_a=*/false, a, /*neg_b=*/false, b,
      /*c_mod=*/(short)0, c, /*reuse_a=*/false, /*reuse_b=*/false);
}

// A-fragment (16x32 f16): lane r=lane&15 holds row M=r; g=lane>>4 selects the
// K-interleave: halves 0..7 = K[8g..8g+7], halves 8..15 = K[16+8g..16+8g+7].
__device__ __forceinline__ v16h load_frag_a(const _Float16* base, int ld,
                                            int r, int g) {
  const _Float16* p = base + (size_t)r * ld + g * 8;
  v8h lo = *(const v8h*)p;
  v8h hi = *(const v8h*)(p + 16);
  v16h f;
#pragma unroll
  for (int i = 0; i < 8; ++i) { f[i] = lo[i]; f[i + 8] = hi[i]; }
  return f;
}

// B-fragment (32x16 f16) from n-major storage Bt[n][k]: lane holds column
// N=lane&15, halves 0..7 = K[8g..8g+7], halves 8..15 = K[16+8g..16+8g+7].
__device__ __forceinline__ v16h load_frag_b(const _Float16* baseNmajor, int ld,
                                            int r, int g) {
  const _Float16* p = baseNmajor + (size_t)r * ld + g * 8;
  v8h lo = *(const v8h*)p;
  v8h hi = *(const v8h*)(p + 16);
  v16h f;
#pragma unroll
  for (int i = 0; i < 8; ++i) { f[i] = lo[i]; f[i + 8] = hi[i]; }
  return f;
}

// ---------------------------------------------------------------------------
// Precision conversion kernels
// ---------------------------------------------------------------------------
__global__ void cvt_f16_kernel(const float* __restrict__ src,
                               _Float16* __restrict__ dst, int n) {
  int i = blockIdx.x * blockDim.x + threadIdx.x;
  if (i < n) dst[i] = (_Float16)src[i];
}

// src[rows][cols] (f32) -> dst[cols][rows] (f16)
__global__ void cvt_transpose_f16_kernel(const float* __restrict__ src,
                                         _Float16* __restrict__ dst,
                                         int rows, int cols) {
  int i = blockIdx.x * blockDim.x + threadIdx.x;
  if (i < rows * cols) {
    int r = i / cols, c = i - r * cols;
    dst[(size_t)c * rows + r] = (_Float16)src[i];
  }
}

// ---------------------------------------------------------------------------
// WMMA GEMM: C[M][N] = A[M][K] * Bt[N][K]^T + bias
// Block tile 128x128, 8 waves (4 in M x 2 in N), wave tile 32x64, K-step 32.
// Tiles staged via TDM (wave 0 issues two descriptors) or async-to-LDS.
// MODE 0: scatter f16 into Q/K/V [B,H,SEQ,HDIM]   MODE 1: f32 into out.
// ---------------------------------------------------------------------------
#define GEMM_LDT 40  // 32 halves + 8 pad; 80 B row stride (16B aligned)

__device__ __forceinline__ void stage_tile_async(_Float16* dst,
                                                 const _Float16* src, int ld,
                                                 int row0, int k0, int tid) {
  // 128 rows x 32 halves (64B) = 512 chunks of 16 bytes; 256 threads -> 2 each
  for (int c = tid; c < 512; c += 256) {
    int r = c >> 2;
    int x = c & 3;
    async_copy16(&dst[r * GEMM_LDT + x * 8],
                 &src[(size_t)(row0 + r) * ld + k0 + x * 8]);
  }
}

template <int MODE>
__global__ __launch_bounds__(256) void gemm_wmma_kernel(
    const _Float16* __restrict__ A,   // [M][K] row-major
    const _Float16* __restrict__ Bt,  // [N][K] row-major (B transposed)
    const float* __restrict__ bias,   // [N]
    float* __restrict__ outF32,       // MODE 1
    _Float16* __restrict__ Qo, _Float16* __restrict__ Ko,
    _Float16* __restrict__ Vo,        // MODE 0
    int M, int N, int K) {
  __shared__ _Float16 sA[2][128 * GEMM_LDT];
  __shared__ _Float16 sB[2][128 * GEMM_LDT];

  const int tid = threadIdx.x;
  const int lane = tid & 31;
  const int wave = tid >> 5;
  const int wm = wave & 3;   // 0..3 -> +32*wm rows
  const int wn = wave >> 2;  // 0..1 -> +64*wn cols
  const int r = lane & 15;
  const int g = lane >> 4;
  const int mBlock = blockIdx.y * 128;
  const int nBlock = blockIdx.x * 128;

  v8f acc[2][4];
#pragma unroll
  for (int tm = 0; tm < 2; ++tm)
#pragma unroll
    for (int tn = 0; tn < 4; ++tn) acc[tm][tn] = (v8f){0, 0, 0, 0, 0, 0, 0, 0};

  auto stage_both = [&](int bufIdx, int k0) {
#if HAVE_TDM
    if (wave == 0) {
      tdm_load_2d(&A[(size_t)mBlock * K + k0], &sA[bufIdx][0], 32, 128, K, true);
      tdm_load_2d(&Bt[(size_t)nBlock * K + k0], &sB[bufIdx][0], 32, 128, K, true);
    }
#else
    stage_tile_async(&sA[bufIdx][0], A, K, mBlock, k0, tid);
    stage_tile_async(&sB[bufIdx][0], Bt, K, nBlock, k0, tid);
#endif
  };

  stage_both(0, 0);
  wait_tile_stage();
  __syncthreads();

  int buf = 0;
  for (int kk = 0; kk < K; kk += 32) {
    if (kk + 32 < K) stage_both(buf ^ 1, kk + 32);

    v16h aF[2], bF[4];
#pragma unroll
    for (int tm = 0; tm < 2; ++tm)
      aF[tm] = load_frag_a(&sA[buf][(wm * 32 + tm * 16) * GEMM_LDT], GEMM_LDT, r, g);
#pragma unroll
    for (int tn = 0; tn < 4; ++tn)
      bF[tn] = load_frag_b(&sB[buf][(wn * 64 + tn * 16) * GEMM_LDT], GEMM_LDT, r, g);
#pragma unroll
    for (int tm = 0; tm < 2; ++tm)
#pragma unroll
      for (int tn = 0; tn < 4; ++tn)
        acc[tm][tn] = wmma16(aF[tm], bF[tn], acc[tm][tn]);

    wait_tile_stage();
    __syncthreads();
    buf ^= 1;
  }

  // Epilogue: C layout -> row = j + 8*g, col = lane&15 within each 16x16 tile
#pragma unroll
  for (int tm = 0; tm < 2; ++tm) {
#pragma unroll
    for (int tn = 0; tn < 4; ++tn) {
      const int col = nBlock + wn * 64 + tn * 16 + r;
      const float bv = bias[col];
#pragma unroll
      for (int j = 0; j < 8; ++j) {
        const int row = mBlock + wm * 32 + tm * 16 + j + 8 * g;
        const float v = acc[tm][tn][j] + bv;
        if (MODE == 1) {
          outF32[(size_t)row * N + col] = v;
        } else {
          const int s = col >> 10;           // 0=q 1=k 2=v
          const int rem = col & 1023;
          const int h = rem >> 6;
          const int d = rem & 63;
          const int b = row >> 11;           // /SEQ
          const int n = row & 2047;
          const size_t off =
              (((size_t)(b * NHEAD + h)) * SEQ + n) * HDIM + d;
          const _Float16 hv = (_Float16)v;
          if (s == 0)      Qo[off] = hv;
          else if (s == 1) Ko[off] = hv;
          else             Vo[off] = hv;
        }
      }
    }
  }
}

// ---------------------------------------------------------------------------
// Flash attention: grid (SEQ/128, B*H). 8 waves; each wave owns 16 q-rows.
// Per kv-block of 64: S = Q K^T (WMMA), online softmax, O += P V (WMMA).
// K block staged to LDS via async-to-LDS (row-major == B-fragment layout);
// V transposed into LDS; P restaged via per-wave LDS (C- -> A-layout).
// O written as [B*SEQ][DMODEL] f16 (row = b*SEQ+n, col = h*64+d).
// ---------------------------------------------------------------------------
__global__ __launch_bounds__(256) void attn_wmma_kernel(
    const _Float16* __restrict__ Q, const _Float16* __restrict__ Km,
    const _Float16* __restrict__ V, _Float16* __restrict__ O) {
  __shared__ _Float16 sK[64][72];       // K rows (kv-major), 144 B rows
  __shared__ _Float16 sVt[64][72];      // Vt[d][kv]
  __shared__ _Float16 sP[8][16][72];    // per-wave P tile 16x64 (+pad)

  const int tid = threadIdx.x;
  const int lane = tid & 31;
  const int wave = tid >> 5;
  const int r = lane & 15;
  const int g = lane >> 4;
  const int bh = blockIdx.y;            // b*NHEAD + h
  const int qBase = blockIdx.x * 128 + wave * 16;
  const float scale = 0.125f;           // 1/sqrt(64)

  // Q fragments for the whole pass (d = 0..31, 32..63)
  const _Float16* Qb = Q + ((size_t)bh * SEQ + qBase) * HDIM;
  v16h qF[2];
#pragma unroll
  for (int ks = 0; ks < 2; ++ks) qF[ks] = load_frag_a(Qb + ks * 32, HDIM, r, g);

  float rowmax[8], rowsum[8];
  v8f oAcc[4];
#pragma unroll
  for (int j = 0; j < 8; ++j) { rowmax[j] = -3.0e38f; rowsum[j] = 0.f; }
#pragma unroll
  for (int dt = 0; dt < 4; ++dt) oAcc[dt] = (v8f){0, 0, 0, 0, 0, 0, 0, 0};

  for (int kv0 = 0; kv0 < SEQ; kv0 += 64) {
    const _Float16* Kb = Km + ((size_t)bh * SEQ + kv0) * HDIM;
    const _Float16* Vb = V + ((size_t)bh * SEQ + kv0) * HDIM;

    // ---- async K staging: 64 rows x 64B = 256 chunks, one per thread
    {
      int kr = tid >> 2, x = tid & 3;
      async_copy16(&sK[kr][x * 8], &Kb[kr * HDIM + x * 8]);
    }
    // ---- cooperative V^T staging: sVt[d][kv] = V[kv0+kv][d]
    for (int idx = tid; idx < 64 * 64; idx += 256) {
      int kv = idx >> 6, d = idx & 63;
      sVt[d][kv] = Vb[kv * HDIM + d];
    }
    // prefetch next kv block (lowers to global_prefetch_b8)
    if (kv0 + 64 < SEQ) {
      __builtin_prefetch(Kb + 64 * HDIM + lane * 64, 0, 0);
      __builtin_prefetch(Vb + 64 * HDIM + lane * 64, 0, 0);
    }
    wait_async_copies();
    __syncthreads();

    // ---- S = (Q K^T) * scale : 4 n-tiles x 2 k-steps, B-frags from LDS
    v8f sAcc[4];
#pragma unroll
    for (int nt = 0; nt < 4; ++nt) {
      v8f c = (v8f){0, 0, 0, 0, 0, 0, 0, 0};
#pragma unroll
      for (int ks = 0; ks < 2; ++ks) {
        v16h bF = load_frag_b(&sK[nt * 16][ks * 32], 72, r, g);
        c = wmma16(qF[ks], bF, c);
      }
#pragma unroll
      for (int j = 0; j < 8; ++j) c[j] *= scale;
      sAcc[nt] = c;
    }

    // ---- online softmax (row j+8g spans the 16 lanes of each half-group)
    float mNew[8], corr[8];
#pragma unroll
    for (int j = 0; j < 8; ++j) {
      float m = fmaxf(fmaxf(sAcc[0][j], sAcc[1][j]),
                      fmaxf(sAcc[2][j], sAcc[3][j]));
#pragma unroll
      for (int s = 1; s < 16; s <<= 1) m = fmaxf(m, __shfl_xor(m, s, 32));
      mNew[j] = fmaxf(rowmax[j], m);
      corr[j] = __expf(rowmax[j] - mNew[j]);
      rowmax[j] = mNew[j];
      rowsum[j] *= corr[j];
    }
#pragma unroll
    for (int dt = 0; dt < 4; ++dt)
#pragma unroll
      for (int j = 0; j < 8; ++j) oAcc[dt][j] *= corr[j];

    // ---- P = exp(S - m), stage to LDS (C-layout write), accumulate row sums
#pragma unroll
    for (int j = 0; j < 8; ++j) {
      float ps = 0.f;
#pragma unroll
      for (int nt = 0; nt < 4; ++nt) {
        float p = __expf(sAcc[nt][j] - mNew[j]);
        ps += p;
        sP[wave][j + 8 * g][nt * 16 + r] = (_Float16)p;
      }
#pragma unroll
      for (int s = 1; s < 16; s <<= 1) ps += __shfl_xor(ps, s, 32);
      rowsum[j] += ps;
    }

    // ---- O += P V : A from per-wave sP, B from shared sVt
#pragma unroll
    for (int ks = 0; ks < 2; ++ks) {
      v16h aF = load_frag_a(&sP[wave][0][ks * 32], 72, r, g);
#pragma unroll
      for (int dt = 0; dt < 4; ++dt) {
        v16h bF = load_frag_b(&sVt[dt * 16][ks * 32], 72, r, g);
        oAcc[dt] = wmma16(aF, bF, oAcc[dt]);
      }
    }
    __syncthreads();
  }

  // ---- normalize and write O[b*SEQ+q][h*64+d] (f16)
  const int b = bh >> 4;
  const int h = bh & 15;
#pragma unroll
  for (int j = 0; j < 8; ++j) {
    const float inv = 1.0f / rowsum[j];
    const int qrow = qBase + j + 8 * g;
    const size_t base = ((size_t)b * SEQ + qrow) * DMODEL + h * HDIM;
#pragma unroll
    for (int dt = 0; dt < 4; ++dt)
      O[base + dt * 16 + r] = (_Float16)(oAcc[dt][j] * inv);
  }
}

// ---------------------------------------------------------------------------
// Launch
// ---------------------------------------------------------------------------
extern "C" void kernel_launch(void* const* d_in, const int* in_sizes, int n_in,
                              void* d_out, int out_size, void* d_ws,
                              size_t ws_size, hipStream_t stream) {
  const float* x     = (const float*)d_in[0];  // [4,2048,1024]
  const float* W_qkv = (const float*)d_in[1];  // [1024,3072]
  const float* b_qkv = (const float*)d_in[2];  // [3072]
  const float* W_out = (const float*)d_in[3];  // [1024,1024]
  const float* b_out = (const float*)d_in[4];  // [1024]
  float* out = (float*)d_out;                  // [4,2048,1024] f32

  char* ws = (char*)d_ws;
  size_t off = 0;
  auto carve = [&](size_t bytes) -> void* {
    void* p = ws + off;
    off += (bytes + 255) & ~(size_t)255;
    return p;
  };
  _Float16* xh    = (_Float16*)carve((size_t)MROWS * DMODEL * 2);       // 16 MB
  _Float16* WqkvT = (_Float16*)carve((size_t)3 * DMODEL * DMODEL * 2);  // 6 MB
  _Float16* WoutT = (_Float16*)carve((size_t)DMODEL * DMODEL * 2);      // 2 MB
  _Float16* Qb    = (_Float16*)carve((size_t)BATCH * NHEAD * SEQ * HDIM * 2);
  _Float16* Kb    = (_Float16*)carve((size_t)BATCH * NHEAD * SEQ * HDIM * 2);
  _Float16* Vb    = (_Float16*)carve((size_t)BATCH * NHEAD * SEQ * HDIM * 2);
  _Float16* Ob    = (_Float16*)carve((size_t)MROWS * DMODEL * 2);       // 16 MB

  // 1) fp32 -> fp16 conversions (weights transposed to n-major for WMMA B)
  {
    int n = MROWS * DMODEL;
    cvt_f16_kernel<<<(n + 255) / 256, 256, 0, stream>>>(x, xh, n);
  }
  {
    int n = DMODEL * 3 * DMODEL;
    cvt_transpose_f16_kernel<<<(n + 255) / 256, 256, 0, stream>>>(
        W_qkv, WqkvT, DMODEL, 3 * DMODEL);
  }
  {
    int n = DMODEL * DMODEL;
    cvt_transpose_f16_kernel<<<(n + 255) / 256, 256, 0, stream>>>(
        W_out, WoutT, DMODEL, DMODEL);
  }

  // 2) fused QKV projection GEMM -> Q/K/V [B,H,SEQ,64] f16
  gemm_wmma_kernel<0><<<dim3(3 * DMODEL / 128, MROWS / 128), 256, 0, stream>>>(
      xh, WqkvT, b_qkv, nullptr, Qb, Kb, Vb, MROWS, 3 * DMODEL, DMODEL);

  // 3) flash attention -> Ob [B*SEQ][DMODEL] f16
  attn_wmma_kernel<<<dim3(SEQ / 128, BATCH * NHEAD), 256, 0, stream>>>(
      Qb, Kb, Vb, Ob);

  // 4) output projection GEMM -> out f32
  gemm_wmma_kernel<1><<<dim3(DMODEL / 128, MROWS / 128), 256, 0, stream>>>(
      Ob, WoutT, b_out, out, nullptr, nullptr, nullptr, MROWS, DMODEL, DMODEL);
}
